// Set2SetThenCat_9122510537146
// MI455X (gfx1250) — compile-verified, matrix operand order checked by reading, not run on hardware
//
#include <hip/hip_runtime.h>
#include <hip/hip_bf16.h>

// ---------- types for WMMA ----------
typedef __attribute__((ext_vector_type(16))) _Float16 v16h;
typedef __attribute__((ext_vector_type(8)))  _Float16 h8_t;
typedef __attribute__((ext_vector_type(8)))  float    v8f;

#define B_GRAPHS 1024
#define D_FEAT   256
#define DG_FEAT  128
#define K_CAT    768     // 2D + D
#define N_GATES  1024    // 4D
#define OUT_W    1152    // 2D + 2D + DG

// ============================================================
// Prep: Wcat (f16, [1024 x 768]) = [W_ih | W_hh], bias = b_ih + b_hh
// ============================================================
__global__ void prep_weights_kernel(const float* __restrict__ Wih,
                                    const float* __restrict__ Whh,
                                    const float* __restrict__ bih,
                                    const float* __restrict__ bhh,
                                    _Float16* __restrict__ Wcat,
                                    float* __restrict__ bias) {
    int n = blockIdx.x;  // 0..1023 gate row
    for (int k = threadIdx.x; k < 512; k += 256)
        Wcat[n * K_CAT + k] = (_Float16)Wih[n * 512 + k];
    for (int k = threadIdx.x; k < 256; k += 256)
        Wcat[n * K_CAT + 512 + k] = (_Float16)Whh[n * 256 + k];
    if (threadIdx.x == 0) bias[n] = bih[n] + bhh[n];
}

__global__ void zero_f32_kernel(float* __restrict__ p, int n) {
    int i = blockIdx.x * 256 + threadIdx.x;
    if (i < n) p[i] = 0.0f;
}

// ============================================================
// LSTM gate GEMM: gates[1024 x 1024] = X[1024 x 768] * Wcat^T + bias
// One wave32 per 16x16 output tile; v_wmma_f32_16x16x32_f16 over K.
// ============================================================
__global__ __launch_bounds__(128)
void lstm_gemm_kernel(const _Float16* __restrict__ X,
                      const _Float16* __restrict__ W,
                      const float* __restrict__ bias,
                      float* __restrict__ gates) {
    const int wave = threadIdx.x >> 5;
    const int lane = threadIdx.x & 31;
    const int tile = blockIdx.x * 4 + wave;   // 4096 tiles total
    const int mt = tile >> 6;                 // 64 m-tiles
    const int nt = tile & 63;                 // 64 n-tiles

    const int m = (mt << 4) + (lane & 15);
    const int n = (nt << 4) + (lane & 15);
    const int kb = (lane < 16) ? 0 : 8;       // ISA 16-bit A/B fragment layout

    const _Float16* xrow = X + (size_t)m * K_CAT + kb;
    const _Float16* wrow = W + (size_t)n * K_CAT + kb;

    v8f acc = {};
    for (int k0 = 0; k0 < K_CAT; k0 += 32) {
        if (k0 + 128 < K_CAT) {               // gfx1250 global_prefetch_b8 path
            __builtin_prefetch(xrow + k0 + 128, 0, 1);
            __builtin_prefetch(wrow + k0 + 128, 0, 1);
        }
        h8_t alo = *(const h8_t*)(xrow + k0);
        h8_t ahi = *(const h8_t*)(xrow + k0 + 16);
        h8_t blo = *(const h8_t*)(wrow + k0);
        h8_t bhi = *(const h8_t*)(wrow + k0 + 16);
        v16h a, b;
#pragma unroll
        for (int i = 0; i < 8; ++i) {
            a[i] = alo[i]; a[i + 8] = ahi[i];
            b[i] = blo[i]; b[i + 8] = bhi[i];
        }
        acc = __builtin_amdgcn_wmma_f32_16x16x32_f16(
            /*neg_a=*/false, a, /*neg_b=*/false, b,
            /*c_mod=*/(short)0, acc, /*reuse_a=*/false, /*reuse_b=*/false);
    }

    // D layout: element (M = mrow0+v, N = n), lanes 16-31 hold M+8
    const int mrow0 = (mt << 4) + ((lane >> 4) << 3);
    const float bn = bias[n];
#pragma unroll
    for (int v = 0; v < 8; ++v) {
        gates[(size_t)(mrow0 + v) * N_GATES + n] = acc[v] + bn;
    }
}

// ============================================================
// LSTM elementwise: gate order i,f,g,o. Updates h,c; writes q (=h)
// into q_star[:, :256] (f32) and x cols [0,256) and [512,768) (f16).
// ============================================================
__global__ __launch_bounds__(256)
void lstm_elem_kernel(const float* __restrict__ gates,
                      float* __restrict__ h, float* __restrict__ c,
                      float* __restrict__ qstar, _Float16* __restrict__ x) {
    const int b = blockIdx.x;
    const int j = threadIdx.x;
    const float* g = gates + (size_t)b * N_GATES;
    float gi = g[j], gf = g[256 + j], gg = g[512 + j], go = g[768 + j];
    float si = 1.0f / (1.0f + __expf(-gi));
    float sf = 1.0f / (1.0f + __expf(-gf));
    float so = 1.0f / (1.0f + __expf(-go));
    float cc = sf * c[b * 256 + j] + si * tanhf(gg);
    float hh = so * tanhf(cc);
    c[b * 256 + j] = cc;
    h[b * 256 + j] = hh;
    qstar[(size_t)b * 512 + j] = hh;
    x[(size_t)b * K_CAT + j] = (_Float16)hh;        // q_star[:, :D]
    x[(size_t)b * K_CAT + 512 + j] = (_Float16)hh;  // h input to next LSTM
}

// ============================================================
// Per-graph attention + softmax + readout. One workgroup per graph.
// feat tile (NPG x 256 f32) staged straight into LDS with the gfx1250
// async path (GLOBAL_LOAD_ASYNC_TO_LDS_B128, ASYNCcnt-tracked); logits,
// segment softmax, and weighted readout all run out of LDS.
// ============================================================
template <int NPG>
__global__ __launch_bounds__(256)
void attention_kernel(const float* __restrict__ feat,
                      const float* __restrict__ h,
                      float* __restrict__ qstar,
                      _Float16* __restrict__ x) {
    extern __shared__ float smem[];
    float* sfeat = smem;               // NPG * 256
    float* sq    = smem + NPG * 256;   // 256
    float* se    = sq + 256;           // NPG (logits, then alpha)

    const int b = blockIdx.x;
    const float* fg = feat + (size_t)b * NPG * D_FEAT;

    // --- async global -> LDS staging (16B per lane per issue) ---
    // GVS addressing: 64-bit base in SGPR pair, per-lane 32-bit byte offset.
    // LDS byte address = low 32 bits of the generic shared-pointer (ISA §10.2).
    {
        const uint32_t lds_base = (uint32_t)(uintptr_t)sfeat;
        for (int i = threadIdx.x; i < NPG * (D_FEAT / 4); i += 256) {
            uint32_t lds_addr = lds_base + (uint32_t)i * 16u;
            uint32_t voff = (uint32_t)i * 16u;
            asm volatile("global_load_async_to_lds_b128 %0, %1, %2"
                         :
                         : "v"(lds_addr), "v"(voff), "s"(fg)
                         : "memory");
        }
        asm volatile("s_wait_asynccnt 0" ::: "memory");
    }
    sq[threadIdx.x] = h[(size_t)b * 256 + threadIdx.x];
    __syncthreads();

    const int wave = threadIdx.x >> 5;
    const int lane = threadIdx.x & 31;

    // logits: e_i = feat_i . q  (one row per wave, wave32 reduce)
    for (int i = wave; i < NPG; i += 8) {
        float p = 0.0f;
#pragma unroll
        for (int t = 0; t < 8; ++t) {
            int j = lane + t * 32;
            p += sfeat[i * D_FEAT + j] * sq[j];
        }
#pragma unroll
        for (int off = 16; off; off >>= 1) p += __shfl_xor(p, off, 32);
        if (lane == 0) se[i] = p;
    }
    __syncthreads();

    // segment softmax (single wave, NPG <= 128)
    if (threadIdx.x < 32) {
        float m = -3.4e38f;
        for (int i = lane; i < NPG; i += 32) m = fmaxf(m, se[i]);
#pragma unroll
        for (int off = 16; off; off >>= 1) m = fmaxf(m, __shfl_xor(m, off, 32));
        float ex[4];
        float s = 0.0f;
        int cnt = 0;
        for (int i = lane; i < NPG; i += 32) {
            float e = __expf(se[i] - m);
            ex[cnt++] = e;
            s += e;
        }
#pragma unroll
        for (int off = 16; off; off >>= 1) s += __shfl_xor(s, off, 32);
        float inv = 1.0f / s;
        cnt = 0;
        for (int i = lane; i < NPG; i += 32) se[i] = ex[cnt++] * inv;
    }
    __syncthreads();

    // readout: r_j = sum_i alpha_i * feat[i][j]
    const int j = threadIdx.x;
    float r = 0.0f;
    for (int i = 0; i < NPG; ++i) r += se[i] * sfeat[i * D_FEAT + j];
    qstar[(size_t)b * 512 + 256 + j] = r;
    x[(size_t)b * K_CAT + 256 + j] = (_Float16)r;   // q_star[:, D:2D]
}

// ============================================================
// Final concat: out[b] = [qa_star (512) | qb_star (512) | feat_global (128)]
// ============================================================
__global__ __launch_bounds__(256)
void final_cat_kernel(const float* __restrict__ qsa,
                      const float* __restrict__ qsb,
                      const float* __restrict__ fglob,
                      float* __restrict__ out) {
    const int b = blockIdx.x;
    for (int j = threadIdx.x; j < OUT_W; j += 256) {
        float v;
        if (j < 512)       v = qsa[(size_t)b * 512 + j];
        else if (j < 1024) v = qsb[(size_t)b * 512 + (j - 512)];
        else               v = fglob[(size_t)b * DG_FEAT + (j - 1024)];
        out[(size_t)b * OUT_W + j] = v;
    }
}

// ============================================================
// Host launcher
// ============================================================
struct ModState {
    _Float16* Wcat;
    float*    bias;
    _Float16* x;       // [1024 x 768] f16 = [q, r, h]
    float*    h;       // [1024 x 256]
    float*    c;       // [1024 x 256]
    float*    qstar;   // [1024 x 512]
    float*    gates;   // [1024 x 1024]
};

static inline size_t align256(size_t v) { return (v + 255) & ~(size_t)255; }

extern "C" void kernel_launch(void* const* d_in, const int* in_sizes, int n_in,
                              void* d_out, int out_size, void* d_ws, size_t ws_size,
                              hipStream_t stream) {
    const float* feat_atom   = (const float*)d_in[0];
    const float* feat_bond   = (const float*)d_in[2];
    const float* feat_global = (const float*)d_in[4];
    const float* Wih[2] = {(const float*)d_in[5], (const float*)d_in[9]};
    const float* Whh[2] = {(const float*)d_in[6], (const float*)d_in[10]};
    const float* bih[2] = {(const float*)d_in[7], (const float*)d_in[11]};
    const float* bhh[2] = {(const float*)d_in[8], (const float*)d_in[12]};

    char* ws = (char*)d_ws;
    size_t off = 0;
    auto alloc = [&](size_t bytes) -> char* {
        char* p = ws + off;
        off = align256(off + bytes);
        return p;
    };

    ModState mod[2];
    for (int m = 0; m < 2; ++m) {
        mod[m].Wcat  = (_Float16*)alloc((size_t)N_GATES * K_CAT * sizeof(_Float16));
        mod[m].bias  = (float*)   alloc((size_t)N_GATES * sizeof(float));
        mod[m].x     = (_Float16*)alloc((size_t)B_GRAPHS * K_CAT * sizeof(_Float16));
        mod[m].h     = (float*)   alloc((size_t)B_GRAPHS * D_FEAT * sizeof(float));
        mod[m].c     = (float*)   alloc((size_t)B_GRAPHS * D_FEAT * sizeof(float));
        mod[m].qstar = (float*)   alloc((size_t)B_GRAPHS * 512 * sizeof(float));
        mod[m].gates = (float*)   alloc((size_t)B_GRAPHS * N_GATES * sizeof(float));
    }

    // ---- prep: f16 weights, combined bias, zero state ----
    for (int m = 0; m < 2; ++m) {
        prep_weights_kernel<<<N_GATES, 256, 0, stream>>>(
            Wih[m], Whh[m], bih[m], bhh[m], mod[m].Wcat, mod[m].bias);
        int nx = B_GRAPHS * K_CAT / 2;  // f16 pairs as f32
        zero_f32_kernel<<<(nx + 255) / 256, 256, 0, stream>>>((float*)mod[m].x, nx);
        int nh = B_GRAPHS * D_FEAT;
        zero_f32_kernel<<<(nh + 255) / 256, 256, 0, stream>>>(mod[m].h, nh);
        zero_f32_kernel<<<(nh + 255) / 256, 256, 0, stream>>>(mod[m].c, nh);
        int nq = B_GRAPHS * 512;
        zero_f32_kernel<<<(nq + 255) / 256, 256, 0, stream>>>(mod[m].qstar, nq);
    }

    // ---- 6 set2set iterations per module ----
    const size_t shmem_atom = (64 * D_FEAT + 256 + 64) * sizeof(float);    // ~66.8 KB
    const size_t shmem_bond = (128 * D_FEAT + 256 + 128) * sizeof(float);  // ~132.6 KB

    for (int it = 0; it < 6; ++it) {
        // atom module
        lstm_gemm_kernel<<<1024, 128, 0, stream>>>(mod[0].x, mod[0].Wcat,
                                                   mod[0].bias, mod[0].gates);
        lstm_elem_kernel<<<B_GRAPHS, 256, 0, stream>>>(mod[0].gates, mod[0].h,
                                                       mod[0].c, mod[0].qstar, mod[0].x);
        attention_kernel<64><<<B_GRAPHS, 256, shmem_atom, stream>>>(
            feat_atom, mod[0].h, mod[0].qstar, mod[0].x);

        // bond module
        lstm_gemm_kernel<<<1024, 128, 0, stream>>>(mod[1].x, mod[1].Wcat,
                                                   mod[1].bias, mod[1].gates);
        lstm_elem_kernel<<<B_GRAPHS, 256, 0, stream>>>(mod[1].gates, mod[1].h,
                                                       mod[1].c, mod[1].qstar, mod[1].x);
        attention_kernel<128><<<B_GRAPHS, 256, shmem_bond, stream>>>(
            feat_bond, mod[1].h, mod[1].qstar, mod[1].x);
    }

    // ---- final concat ----
    final_cat_kernel<<<B_GRAPHS, 256, 0, stream>>>(mod[0].qstar, mod[1].qstar,
                                                   feat_global, (float*)d_out);
}
